// CausalSelfAttention_26654567039173
// MI455X (gfx1250) — matrix operand order repaired and checked by reading, not compiled, not run on hardware
//
#include <hip/hip_runtime.h>

// ---------------------------------------------------------------------------
// CDNA5 (gfx1250) causal self-attention, bf16 WMMA with f32 accumulation.
// GEMMs stage the shared A-tile through LDS via async-to-LDS DMA (ASYNCcnt).
// ---------------------------------------------------------------------------

typedef __attribute__((ext_vector_type(16))) __bf16 v16bf;
typedef __attribute__((ext_vector_type(8)))  float  v8f;

union AFrag { v16bf v; unsigned int u[8]; };
union BFrag { v16bf v; unsigned int u[8]; };
union CFrag { v8f   v; float        f[8]; };

__device__ inline unsigned short f32_to_bf16(float f) {
    unsigned int u = __float_as_uint(f);
    u += 0x7FFFu + ((u >> 16) & 1u);   // round-to-nearest-even
    return (unsigned short)(u >> 16);
}

// A-matrix 16x32 bf16 fragment from row-major [M][ld] (K contiguous).
// ISA layout: lane L: M=L%16, g=L/16; VGPR v, elem e: K = 16*(v/4)+8*g+2*(v%4)+e
static __device__ inline AFrag load_a(const unsigned short* base, int ld, int k0) {
    AFrag a;
    int lane = threadIdx.x & 31;
    int m = lane & 15, g = lane >> 4;
    const unsigned short* row = base + (size_t)m * ld + k0;
#pragma unroll
    for (int v = 0; v < 8; ++v) {
        int k = 16 * (v >> 2) + 8 * g + 2 * (v & 3);
        a.u[v] = *(const unsigned int*)(row + k);
    }
    return a;
}

// B-matrix 32x16 bf16 fragment. Element (K,N) read from row-major [N][ld]
// (K contiguous per N-row). ISA layout: lane L: N=L%16, g=L/16; VGPR v,
// elem e: K = 16*(v/4)+8*g+2*(v%4)+e.
static __device__ inline BFrag load_b(const unsigned short* base, int ld, int k0) {
    BFrag b;
    int lane = threadIdx.x & 31;
    int n = lane & 15, g = lane >> 4;
    const unsigned short* row = base + (size_t)n * ld + k0;
#pragma unroll
    for (int v = 0; v < 8; ++v) b.u[v] = *(const unsigned int*)(row + 16 * (v >> 2) + 8 * g + 2 * (v & 3));
    return b;
}

static __device__ inline v8f wmma_bf16(const AFrag& a, const BFrag& b, v8f c) {
    return __builtin_amdgcn_wmma_f32_16x16x32_bf16(
        false, a.v, false, b.v, (short)0, c, false, false);
}

// ---------------------------------------------------------------------------
// fp32 -> bf16 conversion
// ---------------------------------------------------------------------------
__global__ void cvt_bf16_kernel(const float* __restrict__ src,
                                unsigned short* __restrict__ dst, int n) {
    int i = blockIdx.x * blockDim.x + threadIdx.x;
    if (i < n) dst[i] = f32_to_bf16(src[i]);
}

// ---------------------------------------------------------------------------
// GEMM: out[m][n] = sum_k A[m][k] * W[n][k]   (M=8192, N=K=1024)
// Block = 8 waves sharing one 32-row A tile (async-DMA staged into LDS,
// double buffered); wave w owns a 64-col group.  8 WMMAs / wave / K-step.
// mode 0: Q bf16 [b][h][t][d]    mode 1: K bf16 [b][h][t][d]
// mode 2: V bf16 transposed [b][h][d][t]
// mode 3: fp32 row-major [m][n] (final output)
// ---------------------------------------------------------------------------
__global__ __launch_bounds__(256) void gemm_bf16_kernel(
    const unsigned short* __restrict__ A,   // [8192][1024] bf16
    const unsigned short* __restrict__ W,   // [1024][1024] bf16
    unsigned short* __restrict__ outb,
    float* __restrict__ outf,
    int mode)
{
    const int C = 1024;
    const int NSTEP = C / 32;
    __shared__ unsigned short Alds[2][32 * 32];   // double-buffered A tile

    int lane = threadIdx.x & 31;
    int wvid = threadIdx.x >> 5;      // 0..7
    int mtile = blockIdx.x >> 1;      // 256 row tiles of 32
    int nhalf = blockIdx.x & 1;
    int m0 = mtile * 32;
    int n0 = (nhalf * 8 + wvid) * 64;

    // async-DMA one 32x32 bf16 A tile (2KB) into Alds[buf]: 4 x b128 per lane
    int sr = lane >> 2;               // staging row 0..7
    int sc = (lane & 3) * 16;         // staging byte col 0/16/32/48
    auto stage = [&](int buf, int ks) {
        const unsigned char* g0 =
            (const unsigned char*)(A + (size_t)(m0 + sr) * C + ks * 32) + sc;
        unsigned l0 = (unsigned)(uintptr_t)&Alds[buf][0] + (unsigned)(sr * 64 + sc);
#pragma unroll
        for (int j = 0; j < 4; ++j) {
            unsigned long long g = (unsigned long long)(uintptr_t)g0 +
                                   (unsigned long long)j * 8 * C * 2;
            unsigned l = l0 + (unsigned)(j * 8 * 64);
            asm volatile("global_load_async_to_lds_b128 %0, %1, off"
                         :: "v"(l), "v"(g) : "memory");
        }
    };

    CFrag c[2][4];
#pragma unroll
    for (int i = 0; i < 2; ++i)
#pragma unroll
        for (int j = 0; j < 4; ++j)
            c[i][j].v = (v8f){0.f,0.f,0.f,0.f,0.f,0.f,0.f,0.f};

    if (wvid == 0) stage(0, 0);
    for (int ks = 0; ks < NSTEP; ++ks) {
        int buf = ks & 1;
        if (wvid == 0)
            asm volatile("s_wait_asynccnt 0x0" ::: "memory");  // buf ks landed
        __syncthreads();   // readers released; buf ks^1 fully consumed earlier
        if (wvid == 0 && ks + 1 < NSTEP) stage(buf ^ 1, ks + 1);

        AFrag a0 = load_a(&Alds[buf][0],       32, 0);   // rows m0..m0+15
        AFrag a1 = load_a(&Alds[buf][16 * 32], 32, 0);   // rows m0+16..m0+31
#pragma unroll
        for (int nt = 0; nt < 4; ++nt) {
            BFrag b = load_b(W + (size_t)(n0 + nt * 16) * C, C, ks * 32);
            c[0][nt].v = wmma_bf16(a0, b, c[0][nt].v);
            c[1][nt].v = wmma_bf16(a1, b, c[1][nt].v);
        }
    }

    int nn = lane & 15, hl = lane >> 4;
#pragma unroll
    for (int mt = 0; mt < 2; ++mt)
#pragma unroll
        for (int nt = 0; nt < 4; ++nt)
#pragma unroll
            for (int r = 0; r < 8; ++r) {
                int m = m0 + mt * 16 + 8 * hl + r;   // C/D layout: M = 8*(L/16)+r
                int n = n0 + nt * 16 + nn;           //             N = L%16
                float val = c[mt][nt].f[r];
                if (mode == 3) {
                    outf[(size_t)m * C + n] = val;
                } else {
                    int b = m >> 11, t = m & 2047;
                    int h = n >> 6,  d = n & 63;
                    size_t idx;
                    if (mode == 2) idx = ((size_t)(b * 16 + h) * 64 + d) * 2048 + t;
                    else           idx = ((size_t)(b * 16 + h) * 2048 + t) * 64 + d;
                    outb[idx] = f32_to_bf16(val);
                }
            }
}

// ---------------------------------------------------------------------------
// Flash attention: 1 wave per (b,h,16-query tile). Key blocks of 32.
// Q,K: [bh][2048][64] bf16;  Vt: [bh][64][2048] bf16;  Yb: [b][t][1024] bf16
// ---------------------------------------------------------------------------
__global__ __launch_bounds__(32) void attn_kernel(
    const unsigned short* __restrict__ Q,
    const unsigned short* __restrict__ K,
    const unsigned short* __restrict__ Vt,
    unsigned short* __restrict__ Yb)
{
    __shared__ unsigned short Plds[16 * 32];

    int lane  = threadIdx.x & 31;
    int qtile = blockIdx.x & 127;
    int bh    = blockIdx.x >> 7;          // 0..63
    const unsigned short* Qb = Q  + (size_t)bh * 2048 * 64;
    const unsigned short* Kb = K  + (size_t)bh * 2048 * 64;
    const unsigned short* Vb = Vt + (size_t)bh * 64 * 2048;
    int q0 = qtile * 16;
    int nn = lane & 15, hl = lane >> 4;

    AFrag qa0 = load_a(Qb + (size_t)q0 * 64, 64, 0);   // d = 0..31
    AFrag qa1 = load_a(Qb + (size_t)q0 * 64, 64, 32);  // d = 32..63

    CFrag o[4];
#pragma unroll
    for (int i = 0; i < 4; ++i)
        o[i].v = (v8f){0.f,0.f,0.f,0.f,0.f,0.f,0.f,0.f};
    float m_r[8], l_r[8];
#pragma unroll
    for (int r = 0; r < 8; ++r) { m_r[r] = -1e30f; l_r[r] = 0.f; }

    const v8f zero = (v8f){0.f,0.f,0.f,0.f,0.f,0.f,0.f,0.f};
    int nkb = (q0 >> 5) + 1;              // key blocks up to the diagonal
    for (int kb = 0; kb < nkb; ++kb) {
        int kbase = kb * 32;
        // S = Q . K^T   (two 16x16 tiles, K-dim = 64 split into 2 WMMAs each)
        CFrag s0, s1;
        {
            BFrag b00 = load_b(Kb + (size_t)kbase * 64, 64, 0);
            s0.v = wmma_bf16(qa0, b00, zero);
            BFrag b01 = load_b(Kb + (size_t)kbase * 64, 64, 32);
            s0.v = wmma_bf16(qa1, b01, s0.v);
        }
        {
            BFrag b10 = load_b(Kb + (size_t)(kbase + 16) * 64, 64, 0);
            s1.v = wmma_bf16(qa0, b10, zero);
            BFrag b11 = load_b(Kb + (size_t)(kbase + 16) * 64, 64, 32);
            s1.v = wmma_bf16(qa1, b11, s1.v);
        }
        // scale + causal mask (element (row,col) known from C/D layout)
#pragma unroll
        for (int r = 0; r < 8; ++r) {
            int row = q0 + 8 * hl + r;
            int c0  = kbase + nn;
            int c1  = kbase + 16 + nn;
            float v0 = s0.f[r] * 0.125f;
            float v1 = s1.f[r] * 0.125f;
            s0.f[r] = (c0 <= row) ? v0 : -1e30f;
            s1.f[r] = (c1 <= row) ? v1 : -1e30f;
        }
        // online softmax row stats: reduce across the 16 lanes of each half
        float alpha[8];
#pragma unroll
        for (int r = 0; r < 8; ++r) {
            float mx = fmaxf(s0.f[r], s1.f[r]);
            mx = fmaxf(mx, __shfl_xor(mx, 1, 32));
            mx = fmaxf(mx, __shfl_xor(mx, 2, 32));
            mx = fmaxf(mx, __shfl_xor(mx, 4, 32));
            mx = fmaxf(mx, __shfl_xor(mx, 8, 32));
            float mnew = fmaxf(m_r[r], mx);
            float a = __expf(m_r[r] - mnew);
            alpha[r] = a;
            m_r[r]   = mnew;
            float p0 = __expf(s0.f[r] - mnew);
            float p1 = __expf(s1.f[r] - mnew);
            s0.f[r] = p0; s1.f[r] = p1;
            float sum = p0 + p1;
            sum += __shfl_xor(sum, 1, 32);
            sum += __shfl_xor(sum, 2, 32);
            sum += __shfl_xor(sum, 4, 32);
            sum += __shfl_xor(sum, 8, 32);
            l_r[r] = l_r[r] * a + sum;
        }
        // rescale O accumulators (row of o elem with slot r == row of stats r)
#pragma unroll
        for (int dt = 0; dt < 4; ++dt)
#pragma unroll
            for (int r = 0; r < 8; ++r) o[dt].f[r] *= alpha[r];
        // C/D layout -> A layout via LDS round trip (bf16 P tile, [16][32])
#pragma unroll
        for (int r = 0; r < 8; ++r) {
            int prow = 8 * hl + r;
            Plds[prow * 32 + nn]      = f32_to_bf16(s0.f[r]);
            Plds[prow * 32 + 16 + nn] = f32_to_bf16(s1.f[r]);
        }
        asm volatile("s_wait_dscnt 0x0" ::: "memory");
        AFrag pa = load_a(Plds, 32, 0);
        // O += P . V  (V transposed: rows = d, K-contiguous = t)
#pragma unroll
        for (int dt = 0; dt < 4; ++dt) {
            BFrag vb = load_b(Vb + (size_t)(dt * 16) * 2048, 2048, kbase);
            o[dt].v = wmma_bf16(pa, vb, o[dt].v);
        }
    }

    // normalize and write merged heads: Yb[b][t][h*64 + d]
    int b = bh >> 4, h = bh & 15;
#pragma unroll
    for (int dt = 0; dt < 4; ++dt)
#pragma unroll
        for (int r = 0; r < 8; ++r) {
            int t = q0 + 8 * hl + r;
            float val = o[dt].f[r] / l_r[r];
            Yb[((size_t)b * 2048 + t) * 1024 + h * 64 + dt * 16 + nn] =
                f32_to_bf16(val);
        }
}

// ---------------------------------------------------------------------------
// Launcher
// ---------------------------------------------------------------------------
extern "C" void kernel_launch(void* const* d_in, const int* in_sizes, int n_in,
                              void* d_out, int out_size, void* d_ws, size_t ws_size,
                              hipStream_t stream) {
    const float* x  = (const float*)d_in[0];
    const float* Wq = (const float*)d_in[1];
    const float* Wk = (const float*)d_in[2];
    const float* Wv = (const float*)d_in[3];
    const float* Wo = (const float*)d_in[4];

    const size_t NX = (size_t)8192 * 1024;   // B*T*C elements
    const size_t NW = (size_t)1024 * 1024;   // weight elements

    unsigned short* ws = (unsigned short*)d_ws;
    unsigned short* xb = ws;                 // x  bf16            (NX)
    unsigned short* wb = xb + NX;            // Wq,Wk,Wv,Wo bf16   (4*NW)
    unsigned short* Qb = wb + 4 * NW;        // Q  [bh][t][d]      (NX)
    unsigned short* Kb = Qb + NX;            // K  [bh][t][d]      (NX)
    unsigned short* Vt = Kb + NX;            // V  [bh][d][t]      (NX)
    unsigned short* Yb = Vt + NX;            // y  [b][t][c]       (NX)

    cvt_bf16_kernel<<<(int)((NX + 255) / 256), 256, 0, stream>>>(x,  xb,          (int)NX);
    cvt_bf16_kernel<<<(int)((NW + 255) / 256), 256, 0, stream>>>(Wq, wb + 0 * NW, (int)NW);
    cvt_bf16_kernel<<<(int)((NW + 255) / 256), 256, 0, stream>>>(Wk, wb + 1 * NW, (int)NW);
    cvt_bf16_kernel<<<(int)((NW + 255) / 256), 256, 0, stream>>>(Wv, wb + 2 * NW, (int)NW);
    cvt_bf16_kernel<<<(int)((NW + 255) / 256), 256, 0, stream>>>(Wo, wb + 3 * NW, (int)NW);

    // 512 blocks x 8 waves: block = (m-tile of 32, n-half of 512)
    gemm_bf16_kernel<<<512, 256, 0, stream>>>(xb, wb + 0 * NW, Qb, nullptr, 0);
    gemm_bf16_kernel<<<512, 256, 0, stream>>>(xb, wb + 1 * NW, Kb, nullptr, 1);
    gemm_bf16_kernel<<<512, 256, 0, stream>>>(xb, wb + 2 * NW, Vt, nullptr, 2);

    // one wave per (b,h,qtile): 4*16*128 = 8192 blocks
    attn_kernel<<<8192, 32, 0, stream>>>(Qb, Kb, Vt, Yb);

    gemm_bf16_kernel<<<512, 256, 0, stream>>>(Yb, wb + 3 * NW, nullptr,
                                              (float*)d_out, 3);
}
